// HQLinear_49907519980150
// MI455X (gfx1250) — matrix operand order repaired and checked by reading, not compiled, not run on hardware
//
#include <hip/hip_runtime.h>
#include <hip/hip_bf16.h>

#define IN_F   4096
#define OUT_F  4096
#define CPR    (IN_F / 8)      // codes per output row = 512
#define BM     128
#define BN     128

typedef __attribute__((ext_vector_type(16))) __bf16 v16bf;
typedef __attribute__((ext_vector_type(8)))  __bf16 v8bf;
typedef __attribute__((ext_vector_type(2)))  __bf16 bf16x2;
typedef __attribute__((ext_vector_type(8)))  float  v8f;
typedef __attribute__((ext_vector_type(4)))  int    v4i;

// hardware fp32->bf16 (lowers to v_cvt_pk_bf16_f32 on gfx1250)
static __device__ __forceinline__ unsigned pkc(float a, float b) {
  bf16x2 v = {(__bf16)a, (__bf16)b};
  return __builtin_bit_cast(unsigned, v);
}

// ---------------- 16-byte global->LDS async copy ----------------
static __device__ __forceinline__ void async_cp16(unsigned short* lds,
                                                  const unsigned short* g) {
#if __has_builtin(__builtin_amdgcn_global_load_async_to_lds_b128)
  __builtin_amdgcn_global_load_async_to_lds_b128(
      (__attribute__((address_space(1))) v4i*)(g),
      (__attribute__((address_space(3))) v4i*)(lds),
      0, 0);
#else
  *(uint4*)lds = *(const uint4*)g;
#endif
}
static __device__ __forceinline__ void wait_async_all() {
#if __has_builtin(__builtin_amdgcn_s_wait_asynccnt)
  __builtin_amdgcn_s_wait_asynccnt(0);
#elif __has_builtin(__builtin_amdgcn_global_load_async_to_lds_b128)
  asm volatile("s_wait_asynccnt 0x0" ::: "memory");
#endif
}

// ================= prep pass 1: X fp32 -> bf16 =================
__global__ __launch_bounds__(256)
void cvt_x_bf16(const float* __restrict__ x, unsigned short* __restrict__ xb) {
  const size_t t = (size_t)blockIdx.x * 256 + threadIdx.x;   // 8 elems / thread
  const float4* p = (const float4*)(x + t * 8);
  float4 a = p[0], b = p[1];
  *(uint4*)(xb + t * 8) = make_uint4(pkc(a.x, a.y), pkc(a.z, a.w),
                                     pkc(b.x, b.y), pkc(b.z, b.w));
}

// ============ prep pass 2: dequantize W -> bf16 row-major ============
__global__ __launch_bounds__(256)
void dequant_w_bf16(const int* __restrict__ idx, const float* __restrict__ cbk,
                    const float* __restrict__ scl, unsigned short* __restrict__ wb) {
  const size_t c = (size_t)blockIdx.x * 256 + threadIdx.x;   // one code / thread
  const int o = (int)(c >> 9);                               // c / CPR
  const float s = scl[o];
  const int code = idx[c];
  const float4* cb = (const float4*)(cbk + (size_t)code * 8);
  float4 a = cb[0], b = cb[1];
  *(uint4*)(wb + c * 8) = make_uint4(pkc(a.x * s, a.y * s), pkc(a.z * s, a.w * s),
                                     pkc(b.x * s, b.y * s), pkc(b.z * s, b.w * s));
}

// ================= main GEMM: bf16 WMMA, async double-buffer =================
#define BK2    64
#define NKT2   (IN_F / BK2)    // 64 K-tiles
#define LDSW2  72              // padded row stride (bf16) = 144B, 16B aligned

__global__ __launch_bounds__(256)
void hq_wmma_gemm_async(const unsigned short* __restrict__ xb,
                        const unsigned short* __restrict__ wb,
                        float* __restrict__ out)
{
  __shared__ __align__(16) unsigned short sX[2][BM * LDSW2];
  __shared__ __align__(16) unsigned short sW[2][BN * LDSW2];

  const int tid  = threadIdx.x;
  const int lane = tid & 31;
  const int wv   = tid >> 5;
  const int ls   = lane & 15;
  const int hi   = lane >> 4;

  const int mBase = blockIdx.x * BM;
  const int nBase = blockIdx.y * BN;
  const int wm = (wv & 1) * 64;
  const int wn = (wv >> 1) * 32;

  // tile copy mapping: 128 rows x 64 cols bf16 = 1024 16B-chunks, 4 per thread
  int cpRow[4], cpCol[4];
  const unsigned short* gX[4];
  const unsigned short* gW[4];
#pragma unroll
  for (int i = 0; i < 4; ++i) {
    const int ch = tid + i * 256;
    cpRow[i] = ch >> 3;
    cpCol[i] = (ch & 7) * 8;
    gX[i] = xb + (size_t)(mBase + cpRow[i]) * IN_F + cpCol[i];
    gW[i] = wb + (size_t)(nBase + cpRow[i]) * IN_F + cpCol[i];
  }

  auto issue_async = [&](int b, int kt) {
    const int ko = kt * BK2;
#pragma unroll
    for (int i = 0; i < 4; ++i) {
      async_cp16(&sX[b][cpRow[i] * LDSW2 + cpCol[i]], gX[i] + ko);
      async_cp16(&sW[b][cpRow[i] * LDSW2 + cpCol[i]], gW[i] + ko);
    }
  };

  v8f acc[4][2];
  const v8f vz = {0.f, 0.f, 0.f, 0.f, 0.f, 0.f, 0.f, 0.f};
#pragma unroll
  for (int mi = 0; mi < 4; ++mi)
#pragma unroll
    for (int ni = 0; ni < 2; ++ni) acc[mi][ni] = vz;

  auto compute = [&](int b) {
    const unsigned short* bx = sX[b];
    const unsigned short* bw = sW[b];
#pragma unroll
    for (int s = 0; s < 2; ++s) {          // two K=32 steps per 64-wide tile
      v16bf A[4], B[2];
#pragma unroll
      for (int mi = 0; mi < 4; ++mi) {
        const unsigned short* p = bx + (wm + mi * 16 + ls) * LDSW2 + s * 32 + hi * 8;
        v8bf lo = *(const v8bf*)p;
        v8bf h8 = *(const v8bf*)(p + 16);
        A[mi] = __builtin_shufflevector(lo, h8, 0, 1, 2, 3, 4, 5, 6, 7,
                                                8, 9, 10, 11, 12, 13, 14, 15);
      }
#pragma unroll
      for (int ni = 0; ni < 2; ++ni) {
        const unsigned short* p = bw + (wn + ni * 16 + ls) * LDSW2 + s * 32 + hi * 8;
        v8bf lo = *(const v8bf*)p;
        v8bf h8 = *(const v8bf*)(p + 16);
        B[ni] = __builtin_shufflevector(lo, h8, 0, 1, 2, 3, 4, 5, 6, 7,
                                                8, 9, 10, 11, 12, 13, 14, 15);
      }
#pragma unroll
      for (int mi = 0; mi < 4; ++mi)
#pragma unroll
        for (int ni = 0; ni < 2; ++ni)
          acc[mi][ni] = __builtin_amdgcn_wmma_f32_16x16x32_bf16(
              false, A[mi], false, B[ni], (short)0, acc[mi][ni], false, false);
    }
  };

  issue_async(0, 0);
  wait_async_all();
  __syncthreads();

  int buf = 0;
  for (int kt = 0; kt < NKT2; ++kt) {
    const bool has_next = (kt + 1) < NKT2;
    if (has_next) issue_async(buf ^ 1, kt + 1);   // async engine fetches next tile
    compute(buf);                                  // WMMA on current tile
    if (has_next) wait_async_all();                // own async writes to LDS done
    __syncthreads();                               // everyone's writes visible
    buf ^= 1;
  }

#pragma unroll
  for (int mi = 0; mi < 4; ++mi) {
#pragma unroll
    for (int ni = 0; ni < 2; ++ni) {
      const int n = nBase + wn + ni * 16 + ls;
      float* op = out + (size_t)(mBase + wm + mi * 16 + hi * 8) * OUT_F + n;
#pragma unroll
      for (int r = 0; r < 8; ++r)
        op[(size_t)r * OUT_F] = acc[mi][ni][r];
    }
  }
}

// ================= fallback: fused dequant GEMM (round-1 path) =================
#define BK     32
#define NKT    (IN_F / BK)
#define LDSW   40

__global__ __launch_bounds__(256)
void hq_wmma_gemm_fused(const float* __restrict__ x,
                        const int*   __restrict__ idx,
                        const float* __restrict__ cbk,
                        const float* __restrict__ scl,
                        float*       __restrict__ out)
{
  __shared__ __align__(16) unsigned short sX[2][BM * LDSW];
  __shared__ __align__(16) unsigned short sW[2][BN * LDSW];

  const int tid  = threadIdx.x;
  const int lane = tid & 31;
  const int wv   = tid >> 5;
  const int ls   = lane & 15;
  const int hi   = lane >> 4;

  const int mBase = blockIdx.x * BM;
  const int nBase = blockIdx.y * BN;
  const int wm = (wv & 1) * 64;
  const int wn = (wv >> 1) * 32;

  const int rX = tid >> 1;
  const int cX = (tid & 1) * 16;
  const float* xrow = x + (size_t)(mBase + rX) * IN_F + cX;

  const int rW0 = tid >> 2;
  const int rW1 = rW0 + 64;
  const int jW  = tid & 3;
  const float sc0 = scl[nBase + rW0];
  const float sc1 = scl[nBase + rW1];
  const int* idxp0 = idx + (size_t)(nBase + rW0) * CPR + jW;
  const int* idxp1 = idx + (size_t)(nBase + rW1) * CPR + jW;

  float4 px0, px1, px2, px3;
  float4 c0a, c0b, c1a, c1b;

  auto load_global = [&](int kt) {
    const float* p = xrow + kt * BK;
    px0 = *(const float4*)(p +  0);
    px1 = *(const float4*)(p +  4);
    px2 = *(const float4*)(p +  8);
    px3 = *(const float4*)(p + 12);
    const int kc = kt * (BK / 8);
    const float4* cb0 = (const float4*)(cbk + (size_t)idxp0[kc] * 8);
    const float4* cb1 = (const float4*)(cbk + (size_t)idxp1[kc] * 8);
    c0a = cb0[0]; c0b = cb0[1];
    c1a = cb1[0]; c1b = cb1[1];
  };

  auto store_lds = [&](int b) {
    unsigned short* dx = &sX[b][rX * LDSW + cX];
    *(uint2*)(dx +  0) = make_uint2(pkc(px0.x, px0.y), pkc(px0.z, px0.w));
    *(uint2*)(dx +  4) = make_uint2(pkc(px1.x, px1.y), pkc(px1.z, px1.w));
    *(uint2*)(dx +  8) = make_uint2(pkc(px2.x, px2.y), pkc(px2.z, px2.w));
    *(uint2*)(dx + 12) = make_uint2(pkc(px3.x, px3.y), pkc(px3.z, px3.w));
    *(uint4*)(&sW[b][rW0 * LDSW + jW * 8]) =
        make_uint4(pkc(c0a.x * sc0, c0a.y * sc0), pkc(c0a.z * sc0, c0a.w * sc0),
                   pkc(c0b.x * sc0, c0b.y * sc0), pkc(c0b.z * sc0, c0b.w * sc0));
    *(uint4*)(&sW[b][rW1 * LDSW + jW * 8]) =
        make_uint4(pkc(c1a.x * sc1, c1a.y * sc1), pkc(c1a.z * sc1, c1a.w * sc1),
                   pkc(c1b.x * sc1, c1b.y * sc1), pkc(c1b.z * sc1, c1b.w * sc1));
  };

  v8f acc[4][2];
  const v8f vz = {0.f, 0.f, 0.f, 0.f, 0.f, 0.f, 0.f, 0.f};
#pragma unroll
  for (int mi = 0; mi < 4; ++mi)
#pragma unroll
    for (int ni = 0; ni < 2; ++ni) acc[mi][ni] = vz;

  auto compute = [&](int b) {
    const unsigned short* bx = sX[b];
    const unsigned short* bw = sW[b];
    v16bf A[4], B[2];
#pragma unroll
    for (int mi = 0; mi < 4; ++mi) {
      const unsigned short* p = bx + (wm + mi * 16 + ls) * LDSW + hi * 8;
      v8bf lo = *(const v8bf*)p;
      v8bf h8 = *(const v8bf*)(p + 16);
      A[mi] = __builtin_shufflevector(lo, h8, 0, 1, 2, 3, 4, 5, 6, 7,
                                              8, 9, 10, 11, 12, 13, 14, 15);
    }
#pragma unroll
    for (int ni = 0; ni < 2; ++ni) {
      const unsigned short* p = bw + (wn + ni * 16 + ls) * LDSW + hi * 8;
      v8bf lo = *(const v8bf*)p;
      v8bf h8 = *(const v8bf*)(p + 16);
      B[ni] = __builtin_shufflevector(lo, h8, 0, 1, 2, 3, 4, 5, 6, 7,
                                              8, 9, 10, 11, 12, 13, 14, 15);
    }
#pragma unroll
    for (int mi = 0; mi < 4; ++mi)
#pragma unroll
      for (int ni = 0; ni < 2; ++ni)
        acc[mi][ni] = __builtin_amdgcn_wmma_f32_16x16x32_bf16(
            false, A[mi], false, B[ni], (short)0, acc[mi][ni], false, false);
  };

  load_global(0);
  store_lds(0);
  __syncthreads();

  int buf = 0;
  for (int kt = 0; kt < NKT; ++kt) {
    const bool has_next = (kt + 1) < NKT;
    if (has_next) load_global(kt + 1);
    compute(buf);
    if (has_next) store_lds(buf ^ 1);
    __syncthreads();
    buf ^= 1;
  }

#pragma unroll
  for (int mi = 0; mi < 4; ++mi) {
#pragma unroll
    for (int ni = 0; ni < 2; ++ni) {
      const int n = nBase + wn + ni * 16 + ls;
      float* op = out + (size_t)(mBase + wm + mi * 16 + hi * 8) * OUT_F + n;
#pragma unroll
      for (int r = 0; r < 8; ++r)
        op[(size_t)r * OUT_F] = acc[mi][ni][r];
    }
  }
}

extern "C" void kernel_launch(void* const* d_in, const int* in_sizes, int n_in,
                              void* d_out, int out_size, void* d_ws, size_t ws_size,
                              hipStream_t stream) {
  (void)in_sizes; (void)n_in; (void)out_size;
  const float* x        = (const float*)d_in[0];
  const int*   indices  = (const int*)d_in[1];
  const float* codebook = (const float*)d_in[2];
  const float* scales   = (const float*)d_in[3];
  float*       out      = (float*)d_out;

  const size_t xbBytes = (size_t)2 * 2048 * IN_F * 2;       // 32 MB bf16 X
  const size_t wbBytes = (size_t)OUT_F * IN_F * 2;          // 32 MB bf16 W
  dim3 grid((2 * 2048) / BM, OUT_F / BN);

  if (ws_size >= xbBytes + wbBytes) {
    unsigned short* xb = (unsigned short*)d_ws;
    unsigned short* wb = (unsigned short*)((char*)d_ws + xbBytes);
    cvt_x_bf16<<<(2 * 2048 * IN_F) / (8 * 256), 256, 0, stream>>>(x, xb);
    dequant_w_bf16<<<(OUT_F * (IN_F / 8)) / 256, 256, 0, stream>>>(indices, codebook,
                                                                   scales, wb);
    hq_wmma_gemm_async<<<grid, 256, 0, stream>>>(xb, wb, out);
  } else {
    hq_wmma_gemm_fused<<<grid, 256, 0, stream>>>(x, indices, codebook, scales, out);
  }
}
